// rgcn_9577777070672
// MI455X (gfx1250) — matrix operand order repaired and checked by reading, not compiled, not run on hardware
//
#include <hip/hip_runtime.h>

#define N_NODES 4096
#define T_STEPS 365
#define F_IN    16
#define H_DIM   32

typedef __attribute__((ext_vector_type(16))) __bf16 v16bf;
typedef __attribute__((ext_vector_type(8)))  __bf16 v8bf;
typedef __attribute__((ext_vector_type(4)))  __bf16 v4bf;
typedef __attribute__((ext_vector_type(8)))  float  v8f;

__device__ __forceinline__ float sigm(float v) { return 1.0f / (1.0f + __expf(-v)); }

// ---------------------------------------------------------------------------
// One-time fp32 -> bf16 conversion of A. 64 MB fp32 in, 32 MB bf16 out;
// the bf16 copy stays resident in the 192 MB L2 across all 365 steps.
// ---------------------------------------------------------------------------
__global__ __launch_bounds__(256) void k_convert_a(const float* __restrict__ A,
                                                   __bf16* __restrict__ Abf) {
  size_t i = ((size_t)blockIdx.x * blockDim.x + threadIdx.x) * 4;
  float4 f = *(const float4*)(A + i);
  v4bf b;
  b[0] = (__bf16)f.x; b[1] = (__bf16)f.y; b[2] = (__bf16)f.z; b[3] = (__bf16)f.w;
  *(v4bf*)(Abf + i) = b;
}

// ---------------------------------------------------------------------------
// Per-step graph GEMM:  G = tanh( A[4096,4096] @ Mt^T[4096,64] )
// Mt is stored transposed (64 rows of length 4096) so both A and B fragments
// are contiguous 16B loads in the documented wave32 VGPR layouts.
// Wave = one 16x32 output tile (half=0 -> Hg cols 0..31, half=1 -> Cg).
// A fragment is reused for both column-halves of the tile (2 WMMAs / chunk).
// ---------------------------------------------------------------------------
__global__ __launch_bounds__(256) void k_graph_gemm(const __bf16* __restrict__ Abf,
                                                    const __bf16* __restrict__ Mt,
                                                    float* __restrict__ G) {
  const int lane = threadIdx.x & 31;
  const int wave = threadIdx.x >> 5;                       // 0..7
  const int half = wave & 1;                               // 0: Hg, 1: Cg
  const int m0   = ((blockIdx.x << 2) + (wave >> 1)) << 4; // 16-row tile base

  // A 16x32 bf16 layout: lanes 0-15 hold K {0..7, 16..23}, lanes 16-31 {8..15, 24..31}
  const int kSelA = (lane >> 4) << 3;   // 0 or 8
  // B 32x16 bf16 layout: lanes 0-15 hold K {0..7, 16..23}, lanes 16-31 {8..15, 24..31}
  const int kSelB = (lane >> 4) << 3;   // 0 or 8

  const __bf16* aRow = Abf + (size_t)(m0 + (lane & 15)) * N_NODES + kSelA;
  const int col = half * 32 + (lane & 15);
  const __bf16* b0 = Mt + (size_t)col        * N_NODES + kSelB;
  const __bf16* b1 = Mt + (size_t)(col + 16) * N_NODES + kSelB;

  v8f acc0 = {}; v8f acc1 = {};
  #pragma unroll 2
  for (int k = 0; k < N_NODES; k += 32) {
    union { v16bf v; v8bf h[2]; } a, fb0, fb1;
    a.h[0]   = *(const v8bf*)(aRow + k);
    a.h[1]   = *(const v8bf*)(aRow + k + 16);
    fb0.h[0] = *(const v8bf*)(b0 + k);
    fb0.h[1] = *(const v8bf*)(b0 + k + 16);
    fb1.h[0] = *(const v8bf*)(b1 + k);
    fb1.h[1] = *(const v8bf*)(b1 + k + 16);
    __builtin_prefetch(aRow + k + 128, 0, 3);   // global_prefetch next A line
    acc0 = __builtin_amdgcn_wmma_f32_16x16x32_bf16(false, a.v, false, fb0.v,
                                                   (short)0, acc0, false, false);
    acc1 = __builtin_amdgcn_wmma_f32_16x16x32_bf16(false, a.v, false, fb1.v,
                                                   (short)0, acc1, false, false);
  }

  // C/D layout: VGPR r -> row m0+r (lanes 0-15) / m0+8+r (lanes 16-31), col = lane&15
  const int rowBase = m0 + ((lane >> 4) << 3);
  const int c0 = half * 32 + (lane & 15);
  #pragma unroll
  for (int r = 0; r < 8; ++r) {
    G[(size_t)(rowBase + r) * 64 + c0]      = tanhf(acc0[r]);
    G[(size_t)(rowBase + r) * 64 + c0 + 16] = tanhf(acc1[r]);
  }
}

// ---------------------------------------------------------------------------
// Fused step tail, lane-parallel: one WAVE per graph row, lane j owns output
// column j of every 32-wide product. Row vectors are broadcast across the
// wave with __shfl (constant source lane -> v_readlane + scalar-operand FMA).
// For step t (t>=0):
//   h_upd = sigmoid(h_cur@W_h_cur + Hg@W_h_prev + b_h)   (Hg = G[:,0:32])
//   c_upd = sigmoid(c_cur@W_c_cur + Cg@W_c_prev + b_c)   (Cg = G[:,32:64])
//   out[t,n] = h_upd . W_out + b_out                     (shfl_xor reduction)
// Then (t+1 < T) it prepares step t+1:
//   Mt = [ (h_upd@W_graph_h + b) | (c_upd@W_graph_c + b) ]^T  (bf16, via LDS)
//   gates = x[:,t+1]@K + h_upd@R + bias -> h_cur, c_cur for step t+1
// t == -1 is the bootstrap (h_upd = c_upd = 0, no fusion/output).
// ---------------------------------------------------------------------------
__global__ __launch_bounds__(256) void k_fuse_prep(
    const float* __restrict__ G,         // [N][64]
    float* __restrict__ h_lstm,          // [N][32]
    float* __restrict__ c_lstm,          // [N][32]
    __bf16* __restrict__ Mt,             // [64][N]
    float* __restrict__ out,             // [T*N]
    const float* __restrict__ x,         // [N][T][F]
    const float* __restrict__ W_out, const float* __restrict__ b_out,
    const float* __restrict__ W_graph_h, const float* __restrict__ b_graph_h,
    const float* __restrict__ W_graph_c, const float* __restrict__ b_graph_c,
    const float* __restrict__ W_h_cur, const float* __restrict__ W_h_prev,
    const float* __restrict__ b_h,
    const float* __restrict__ W_c_cur, const float* __restrict__ W_c_prev,
    const float* __restrict__ b_c,
    const float* __restrict__ lstm_kernel, const float* __restrict__ lstm_rec,
    const float* __restrict__ lstm_bias,
    int t) {
  __shared__ float sWhc[1024], sWhp[1024], sWcc[1024], sWcp[1024], sWgh[1024], sWgc[1024];
  __shared__ float sK[2048], sR[4096];
  __shared__ float sBh[32], sBc[32], sBgh[32], sBgc[32], sBias[128], sWout[32], sBout[1];
  __shared__ __bf16 sMt[64 * 8];        // staging for coalesced Mt stores
  const int tid = threadIdx.x;
  for (int i = tid; i < 1024; i += 256) {
    sWhc[i] = W_h_cur[i];  sWhp[i] = W_h_prev[i];
    sWcc[i] = W_c_cur[i];  sWcp[i] = W_c_prev[i];
    sWgh[i] = W_graph_h[i]; sWgc[i] = W_graph_c[i];
  }
  for (int i = tid; i < 2048; i += 256) sK[i] = lstm_kernel[i];
  for (int i = tid; i < 4096; i += 256) sR[i] = lstm_rec[i];
  if (tid < 128) sBias[tid] = lstm_bias[tid];
  if (tid < 32) {
    sBh[tid] = b_h[tid]; sBc[tid] = b_c[tid];
    sBgh[tid] = b_graph_h[tid]; sBgc[tid] = b_graph_c[tid];
    sWout[tid] = W_out[tid];
  }
  if (tid == 0) sBout[0] = b_out[0];
  __syncthreads();

  const int lane = tid & 31;
  const int w    = tid >> 5;                 // 0..7: row within block
  const int n0   = blockIdx.x * 8;
  const int n    = n0 + w;

  float hupd = 0.0f, cupd = 0.0f;
  if (t >= 0) {
    float hc = h_lstm[n * H_DIM + lane];
    float cc = c_lstm[n * H_DIM + lane];
    float hg = G[(size_t)n * 64 + lane];
    float cg = G[(size_t)n * 64 + 32 + lane];
    float sh = sBh[lane], sc = sBc[lane];
    #pragma unroll
    for (int k = 0; k < 32; ++k) {
      float hck = __shfl(hc, k, 32);
      float hgk = __shfl(hg, k, 32);
      float cck = __shfl(cc, k, 32);
      float cgk = __shfl(cg, k, 32);
      sh += hck * sWhc[k * 32 + lane] + hgk * sWhp[k * 32 + lane];
      sc += cck * sWcc[k * 32 + lane] + cgk * sWcp[k * 32 + lane];
    }
    hupd = sigm(sh);
    cupd = sigm(sc);
    // out[t,n] = h_upd . W_out + b_out  (wave reduction)
    float o = hupd * sWout[lane];
    #pragma unroll
    for (int d = 16; d > 0; d >>= 1) o += __shfl_xor(o, d, 32);
    if (lane == 0) out[(size_t)t * N_NODES + n] = o + sBout[0];
  }

  const int tn = t + 1;
  if (tn < T_STEPS) {
    // Graph projections for step tn (lane j = column j), staged in LDS.
    float sh = sBgh[lane], sc = sBgc[lane];
    #pragma unroll
    for (int k = 0; k < 32; ++k) {
      float hk = __shfl(hupd, k, 32);
      float ck = __shfl(cupd, k, 32);
      sh += hk * sWgh[k * 32 + lane];
      sc += ck * sWgc[k * 32 + lane];
    }
    sMt[lane * 8 + w]        = (__bf16)sh;
    sMt[(32 + lane) * 8 + w] = (__bf16)sc;

    // LSTM cell for step tn (i,f,g,o column order); lane j owns column j.
    float xv = x[((size_t)n * T_STEPS + tn) * F_IN + (lane & 15)];
    float gi = sBias[lane], gf = sBias[32 + lane], gg = sBias[64 + lane], go = sBias[96 + lane];
    #pragma unroll
    for (int f = 0; f < 16; ++f) {
      float xf = __shfl(xv, f, 32);
      gi += xf * sK[f * 128 + lane];
      gf += xf * sK[f * 128 + 32 + lane];
      gg += xf * sK[f * 128 + 64 + lane];
      go += xf * sK[f * 128 + 96 + lane];
    }
    #pragma unroll
    for (int k = 0; k < 32; ++k) {
      float hk = __shfl(hupd, k, 32);
      gi += hk * sR[k * 128 + lane];
      gf += hk * sR[k * 128 + 32 + lane];
      gg += hk * sR[k * 128 + 64 + lane];
      go += hk * sR[k * 128 + 96 + lane];
    }
    float ig = sigm(gi), fg = sigm(gf), g2 = tanhf(gg), og = sigm(go);
    float cn = fg * cupd + ig * g2;
    float hn = og * tanhf(cn);
    c_lstm[n * H_DIM + lane] = cn;
    h_lstm[n * H_DIM + lane] = hn;

    // Coalesced write-out of the staged Mt tile (64 cols x 8 rows, bf16).
    __syncthreads();
    for (int i = tid; i < 512; i += 256) {
      int j = i >> 3, r = i & 7;
      Mt[(size_t)j * N_NODES + n0 + r] = sMt[i];
    }
  }
}

extern "C" void kernel_launch(void* const* d_in, const int* in_sizes, int n_in,
                              void* d_out, int out_size, void* d_ws, size_t ws_size,
                              hipStream_t stream) {
  (void)in_sizes; (void)n_in; (void)out_size; (void)ws_size;
  const float* x           = (const float*)d_in[0];
  const float* A           = (const float*)d_in[1];
  const float* W_out       = (const float*)d_in[2];
  const float* b_out       = (const float*)d_in[3];
  const float* W_graph_h   = (const float*)d_in[4];
  const float* b_graph_h   = (const float*)d_in[5];
  const float* W_graph_c   = (const float*)d_in[6];
  const float* b_graph_c   = (const float*)d_in[7];
  const float* W_h_cur     = (const float*)d_in[8];
  const float* W_h_prev    = (const float*)d_in[9];
  const float* b_h         = (const float*)d_in[10];
  const float* W_c_cur     = (const float*)d_in[11];
  const float* W_c_prev    = (const float*)d_in[12];
  const float* b_c         = (const float*)d_in[13];
  const float* lstm_kernel = (const float*)d_in[14];
  const float* lstm_rec    = (const float*)d_in[15];
  const float* lstm_bias   = (const float*)d_in[16];

  // Workspace layout (needs ~34.5 MB):
  //   Abf  : bf16 [4096][4096]  32 MB   (L2-resident across the scan)
  //   Mt   : bf16 [64][4096]    512 KB  (graph-GEMM B operand, transposed)
  //   G    : f32  [4096][64]    1 MB    (tanh'd graph states)
  //   h/c  : f32  [4096][32]x2  1 MB    (LSTM state)
  char* ws = (char*)d_ws;
  __bf16* Abf   = (__bf16*)(ws);
  __bf16* Mt    = (__bf16*)(ws + (size_t)32 * 1024 * 1024);
  float*  G     = (float*)(ws + (size_t)32 * 1024 * 1024 + 512 * 1024);
  float*  hLstm = (float*)(ws + (size_t)32 * 1024 * 1024 + 512 * 1024 + 1024 * 1024);
  float*  cLstm = hLstm + (size_t)N_NODES * H_DIM;
  float*  out   = (float*)d_out;

  k_convert_a<<<(N_NODES * (N_NODES / 4)) / 256, 256, 0, stream>>>(A, Abf);

  // Bootstrap: h_prev = c_prev = 0 -> produce Mt, h_cur, c_cur for step 0.
  k_fuse_prep<<<N_NODES / 8, 256, 0, stream>>>(
      G, hLstm, cLstm, Mt, out, x,
      W_out, b_out, W_graph_h, b_graph_h, W_graph_c, b_graph_c,
      W_h_cur, W_h_prev, b_h, W_c_cur, W_c_prev, b_c,
      lstm_kernel, lstm_rec, lstm_bias, -1);

  for (int t = 0; t < T_STEPS; ++t) {
    k_graph_gemm<<<N_NODES / 64, 256, 0, stream>>>(Abf, Mt, G);
    k_fuse_prep<<<N_NODES / 8, 256, 0, stream>>>(
        G, hLstm, cLstm, Mt, out, x,
        W_out, b_out, W_graph_h, b_graph_h, W_graph_c, b_graph_c,
        W_h_cur, W_h_prev, b_h, W_c_cur, W_c_prev, b_c,
        lstm_kernel, lstm_rec, lstm_bias, t);
  }
}
